// GATNet_1881195675933
// MI455X (gfx1250) — matrix operand order repaired and checked by reading, not compiled, not run on hardware
//
#include <hip/hip_runtime.h>
#include <hip/hip_bf16.h>

typedef __attribute__((ext_vector_type(16))) _Float16 v16h;
typedef __attribute__((ext_vector_type(8)))  float    v8f;

#define NEG_SLOPE 0.2f

// ---------------------------------------------------------------------------
// Zero fill
// ---------------------------------------------------------------------------
__global__ void fill_zero_kernel(float* __restrict__ p, int n) {
    int i = blockIdx.x * blockDim.x + threadIdx.x;
    if (i < n) p[i] = 0.0f;
}

// ---------------------------------------------------------------------------
// Helpers: convert 8 consecutive floats (two float4) into halves of a v16h
// ---------------------------------------------------------------------------
__device__ __forceinline__ void cvt8(const float* __restrict__ src, v16h& a, int base) {
    const float4* v = (const float4*)src;
    float4 p0 = v[0];
    float4 p1 = v[1];
    a[base + 0] = (_Float16)p0.x; a[base + 1] = (_Float16)p0.y;
    a[base + 2] = (_Float16)p0.z; a[base + 3] = (_Float16)p0.w;
    a[base + 4] = (_Float16)p1.x; a[base + 5] = (_Float16)p1.y;
    a[base + 6] = (_Float16)p1.z; a[base + 7] = (_Float16)p1.w;
}

// ---------------------------------------------------------------------------
// GEMM1: H1[N,64] = X[N,128] @ W1[128,64]   (WMMA f32<-f16, K=128)
// 128 threads = 4 waves; each wave owns a 16-row tile, 4 n-tiles of 16.
// W1 staged in LDS pre-swizzled into B-fragment layout:
//   frag(ks,nt) for lane L = 16 contiguous halves at ((ks*4+nt)*32+L)*16
//   -> one v16h fragment = two ds_load_b128 per WMMA.
// ---------------------------------------------------------------------------
__global__ __launch_bounds__(128) void gemm1_wmma_kernel(
    const float* __restrict__ X, const float* __restrict__ W1,
    float* __restrict__ H1, int Nrows)
{
    __shared__ __align__(32) _Float16 lw[128 * 64];
    int tid = threadIdx.x;
    for (int i = tid; i < 128 * 64; i += 128) {
        int k = i >> 6, n = i & 63;
        int ks = k >> 5, kin = k & 31;
        int lane = ((kin >> 4) << 4) | (n & 15);
        int ii = kin & 15;
        int nt = n >> 4;
        lw[(((ks * 4 + nt) * 32 + lane) << 4) | ii] = (_Float16)W1[i];
    }
    __syncthreads();

    int wave = tid >> 5;
    int lane = tid & 31;
    int hi   = lane >> 4;                  // 0: lanes 0-15, 1: lanes 16-31
    int rowTile = (blockIdx.x * 4 + wave) * 16;

    int lrow = rowTile + (lane & 15);
    if (lrow >= Nrows) lrow = Nrows - 1;   // clamp loads; stores predicated
    const float* xrow = X + (size_t)lrow * 128;

    v8f acc0 = {}, acc1 = {}, acc2 = {}, acc3 = {};
    int ncol = lane & 15;

    for (int ks = 0; ks < 4; ++ks) {       // K = 4 * 32
        // A fragment (16x32 f16): lanes 0-15 K {0-15}, lanes 16-31 K {16-31}
        int ka = ks * 32 + hi * 16;
        v16h a;
        cvt8(xrow + ka,      a, 0);
        cvt8(xrow + ka + 8,  a, 8);

#pragma unroll
        for (int nt = 0; nt < 4; ++nt) {
            v16h b = *(const v16h*)&lw[(((ks * 4 + nt) * 32 + lane) << 4)];
            v8f* accp = (nt == 0) ? &acc0 : (nt == 1) ? &acc1 : (nt == 2) ? &acc2 : &acc3;
            *accp = __builtin_amdgcn_wmma_f32_16x16x32_f16(
                false, a, false, b, (short)0, *accp, false, false);
        }
    }

    // C layout: VGPR r -> row rowTile + r (+8 for lanes 16-31), col = lane&15
    int rbase = rowTile + (hi ? 8 : 0);
    if (rowTile + 16 <= Nrows) {           // uniform fast path: no predication
#pragma unroll
        for (int r = 0; r < 8; ++r) {
            float* orow = H1 + (size_t)(rbase + r) * 64 + ncol;
            orow[0]  = acc0[r];
            orow[16] = acc1[r];
            orow[32] = acc2[r];
            orow[48] = acc3[r];
        }
    } else {
#pragma unroll
        for (int r = 0; r < 8; ++r) {
            int row = rbase + r;
            if (row < Nrows) {
                float* orow = H1 + (size_t)row * 64 + ncol;
                orow[0]  = acc0[r];
                orow[16] = acc1[r];
                orow[32] = acc2[r];
                orow[48] = acc3[r];
            }
        }
    }
}

// ---------------------------------------------------------------------------
// GEMM2: H2[N,40] = ELU_OUT1[N,64] @ W2[64,40]   (K=64, N padded to 48)
// ---------------------------------------------------------------------------
__global__ __launch_bounds__(128) void gemm2_wmma_kernel(
    const float* __restrict__ Hin, const float* __restrict__ W2,
    float* __restrict__ H2, int Nrows)
{
    __shared__ __align__(32) _Float16 lw[64 * 48];
    int tid = threadIdx.x;
    for (int i = tid; i < 64 * 48; i += 128) {
        int k = i / 48, n = i % 48;
        float v = (n < 40) ? W2[k * 40 + n] : 0.0f;
        int ks = k >> 5, kin = k & 31;
        int lane = ((kin >> 4) << 4) | (n & 15);
        int ii = kin & 15;
        int nt = n >> 4;
        lw[(((ks * 3 + nt) * 32 + lane) << 4) | ii] = (_Float16)v;
    }
    __syncthreads();

    int wave = tid >> 5;
    int lane = tid & 31;
    int hi   = lane >> 4;
    int rowTile = (blockIdx.x * 4 + wave) * 16;

    int lrow = rowTile + (lane & 15);
    if (lrow >= Nrows) lrow = Nrows - 1;
    const float* xrow = Hin + (size_t)lrow * 64;

    v8f acc0 = {}, acc1 = {}, acc2 = {};
    int ncol = lane & 15;

    for (int ks = 0; ks < 2; ++ks) {       // K = 2 * 32
        int ka = ks * 32 + hi * 16;
        v16h a;
        cvt8(xrow + ka,      a, 0);
        cvt8(xrow + ka + 8,  a, 8);

#pragma unroll
        for (int nt = 0; nt < 3; ++nt) {
            v16h b = *(const v16h*)&lw[(((ks * 3 + nt) * 32 + lane) << 4)];
            v8f* accp = (nt == 0) ? &acc0 : (nt == 1) ? &acc1 : &acc2;
            *accp = __builtin_amdgcn_wmma_f32_16x16x32_f16(
                false, a, false, b, (short)0, *accp, false, false);
        }
    }

    int rbase = rowTile + (hi ? 8 : 0);
#pragma unroll
    for (int r = 0; r < 8; ++r) {
        int row = rbase + r;
        if (row >= Nrows) continue;
        float* orow = H2 + (size_t)row * 40;
        // nt 0,1: cols 0-31 always valid
        orow[ncol]      = acc0[r];
        orow[16 + ncol] = acc1[r];
        if (ncol < 8) orow[32 + ncol] = acc2[r];   // cols 32-39 only
    }
}

// ---------------------------------------------------------------------------
// Per-node attention dot tables: ai[n,h] = <H1[n,h,:], att_i[h,:]>,
//                                aj[n,h] = <H1[n,h,:], att_j[h,:]>
// att1 flat layout: [h][0:8]=att_i, [h][8:16]=att_j
// ---------------------------------------------------------------------------
__global__ void att1_dots_kernel(const float* __restrict__ H1,
                                 const float* __restrict__ att1,
                                 float* __restrict__ ai, float* __restrict__ aj,
                                 int N)
{
    int n = blockIdx.x * blockDim.x + threadIdx.x;
    if (n >= N) return;
    const float* hr = H1 + (size_t)n * 64;
#pragma unroll
    for (int hd = 0; hd < 8; ++hd) {
        float si = 0.f, sj = 0.f;
#pragma unroll
        for (int c = 0; c < 8; ++c) {
            float v = hr[hd * 8 + c];
            si += v * att1[hd * 16 + c];
            sj += v * att1[hd * 16 + 8 + c];
        }
        ai[n * 8 + hd] = si;
        aj[n * 8 + hd] = sj;
    }
}

// ---------------------------------------------------------------------------
// Layer 1 edge pass A: denom[dst,h] += exp(lrelu(ai[dst,h]+aj[src,h]) - max_h)
// Edge e < E comes from edge_index; e >= E is the self loop (e-E, e-E).
// ---------------------------------------------------------------------------
__global__ void edge1_denom_kernel(const int* __restrict__ ei,
                                   const float* __restrict__ ai,
                                   const float* __restrict__ aj,
                                   float* __restrict__ denom,
                                   int E, int N)
{
    int e = blockIdx.x * blockDim.x + threadIdx.x;
    if (e >= E + N) return;
    if (e + 8192 < E) {                      // stream-ahead on the index arrays
        __builtin_prefetch(&ei[e + 8192], 0, 1);
        __builtin_prefetch(&ei[E + e + 8192], 0, 1);
    }
    int src = (e < E) ? ei[e]     : (e - E);
    int dst = (e < E) ? ei[E + e] : (e - E);

    float al[8];
    float m = -1e30f;
#pragma unroll
    for (int hd = 0; hd < 8; ++hd) {
        float a = ai[dst * 8 + hd] + aj[src * 8 + hd];
        a = (a > 0.f) ? a : NEG_SLOPE * a;
        al[hd] = a;
        m = fmaxf(m, a);
    }
#pragma unroll
    for (int hd = 0; hd < 8; ++hd)
        atomicAdd(&denom[dst * 8 + hd], __expf(al[hd] - m));
}

// ---------------------------------------------------------------------------
// Layer 1 edge pass B: out1[dst,h,c] += H1[src,h,c] * e/denom[dst,h]
// (recompute e instead of spilling 54 MB of per-edge values)
// ---------------------------------------------------------------------------
__global__ void edge1_aggr_kernel(const int* __restrict__ ei,
                                  const float* __restrict__ ai,
                                  const float* __restrict__ aj,
                                  const float* __restrict__ denom,
                                  const float* __restrict__ H1,
                                  float* __restrict__ out1,
                                  int E, int N)
{
    int e = blockIdx.x * blockDim.x + threadIdx.x;
    if (e >= E + N) return;
    if (e + 8192 < E) {
        __builtin_prefetch(&ei[e + 8192], 0, 1);
        __builtin_prefetch(&ei[E + e + 8192], 0, 1);
    }
    int src = (e < E) ? ei[e]     : (e - E);
    int dst = (e < E) ? ei[E + e] : (e - E);

    float al[8];
    float m = -1e30f;
#pragma unroll
    for (int hd = 0; hd < 8; ++hd) {
        float a = ai[dst * 8 + hd] + aj[src * 8 + hd];
        a = (a > 0.f) ? a : NEG_SLOPE * a;
        al[hd] = a;
        m = fmaxf(m, a);
    }
    const float* hs = H1   + (size_t)src * 64;
    float*       od = out1 + (size_t)dst * 64;
#pragma unroll
    for (int hd = 0; hd < 8; ++hd) {
        float w = __expf(al[hd] - m) / denom[dst * 8 + hd];
        float4 v0 = *(const float4*)(hs + hd * 8);
        float4 v1 = *(const float4*)(hs + hd * 8 + 4);
        atomicAdd(&od[hd * 8 + 0], v0.x * w);
        atomicAdd(&od[hd * 8 + 1], v0.y * w);
        atomicAdd(&od[hd * 8 + 2], v0.z * w);
        atomicAdd(&od[hd * 8 + 3], v0.w * w);
        atomicAdd(&od[hd * 8 + 4], v1.x * w);
        atomicAdd(&od[hd * 8 + 5], v1.y * w);
        atomicAdd(&od[hd * 8 + 6], v1.z * w);
        atomicAdd(&od[hd * 8 + 7], v1.w * w);
    }
}

// ---------------------------------------------------------------------------
// ELU in place
// ---------------------------------------------------------------------------
__global__ void elu_kernel(float* __restrict__ p, int n) {
    int i = blockIdx.x * blockDim.x + threadIdx.x;
    if (i >= n) return;
    float v = p[i];
    p[i] = (v > 0.f) ? v : (__expf(v) - 1.0f);
}

// ---------------------------------------------------------------------------
// Layer 2 edge pass. heads==1 => exp(alpha - alpha.max(axis=1)) == 1 exactly,
// so attention reduces to an in-neighbor mean: accumulate sum + degree.
// ---------------------------------------------------------------------------
__global__ void edge2_aggr_kernel(const int* __restrict__ ei,
                                  const float* __restrict__ H2,
                                  float* __restrict__ out2,
                                  float* __restrict__ deg,
                                  int E, int N)
{
    int e = blockIdx.x * blockDim.x + threadIdx.x;
    if (e >= E + N) return;
    if (e + 8192 < E) {
        __builtin_prefetch(&ei[e + 8192], 0, 1);
        __builtin_prefetch(&ei[E + e + 8192], 0, 1);
    }
    int src = (e < E) ? ei[e]     : (e - E);
    int dst = (e < E) ? ei[E + e] : (e - E);

    atomicAdd(&deg[dst], 1.0f);
    const float* hs = H2   + (size_t)src * 40;
    float*       od = out2 + (size_t)dst * 40;
#pragma unroll
    for (int c = 0; c < 40; ++c)
        atomicAdd(&od[c], hs[c]);
}

// ---------------------------------------------------------------------------
// Final: out[n,:] = log_softmax(out2[n,:] / deg[n])
// ---------------------------------------------------------------------------
__global__ void logsoftmax_kernel(const float* __restrict__ out2,
                                  const float* __restrict__ deg,
                                  float* __restrict__ out, int N)
{
    int n = blockIdx.x * blockDim.x + threadIdx.x;
    if (n >= N) return;
    float inv = 1.0f / deg[n];
    float v[40];
    float m = -1e30f;
#pragma unroll
    for (int c = 0; c < 40; ++c) {
        v[c] = out2[(size_t)n * 40 + c] * inv;
        m = fmaxf(m, v[c]);
    }
    float s = 0.f;
#pragma unroll
    for (int c = 0; c < 40; ++c) s += __expf(v[c] - m);
    float lse = m + __logf(s);
#pragma unroll
    for (int c = 0; c < 40; ++c) out[(size_t)n * 40 + c] = v[c] - lse;
}

// ---------------------------------------------------------------------------
// Host launcher
// ---------------------------------------------------------------------------
extern "C" void kernel_launch(void* const* d_in, const int* in_sizes, int n_in,
                              void* d_out, int out_size, void* d_ws, size_t ws_size,
                              hipStream_t stream)
{
    const float* x    = (const float*)d_in[0];   // [N,128]
    const int*   ei   = (const int*)  d_in[1];   // [2,E] flat
    const float* W1   = (const float*)d_in[2];   // [128,64]
    const float* att1 = (const float*)d_in[3];   // [1,8,16]
    const float* W2   = (const float*)d_in[4];   // [64,40]
    // d_in[5] (att2) provably has no effect: heads==1 => exp(a - a.max) == 1.

    const int N = in_sizes[0] / 128;             // 100000
    const int E = in_sizes[1] / 2;               // 1600000
    const int ET = E + N;                        // edges incl. self loops

    char* ws = (char*)d_ws;
    float* H1    = (float*)ws;                                  // N*64
    float* out1  = (float*)(ws + (size_t)N * 64 * 4);           // N*64
    float* ai    = (float*)(ws + (size_t)N * 128 * 4);          // N*8
    float* aj    = ai + (size_t)N * 8;                          // N*8
    float* denom = aj + (size_t)N * 8;                          // N*8
    float* H2    = denom + (size_t)N * 8;                       // N*40
    float* out2  = H1;     // reuse: H1 is dead after edge1_aggr
    float* deg   = ai;     // reuse: ai is dead after edge1_aggr

    const int TB = 256;
    dim3 b256(TB);

    // Layer 1
    gemm1_wmma_kernel<<<dim3((N + 63) / 64), dim3(128), 0, stream>>>(x, W1, H1, N);
    att1_dots_kernel <<<dim3((N + TB - 1) / TB), b256, 0, stream>>>(H1, att1, ai, aj, N);
    fill_zero_kernel <<<dim3((N * 8 + TB - 1) / TB), b256, 0, stream>>>(denom, N * 8);
    fill_zero_kernel <<<dim3((N * 64 + TB - 1) / TB), b256, 0, stream>>>(out1, N * 64);
    edge1_denom_kernel<<<dim3((ET + TB - 1) / TB), b256, 0, stream>>>(ei, ai, aj, denom, E, N);
    edge1_aggr_kernel <<<dim3((ET + TB - 1) / TB), b256, 0, stream>>>(ei, ai, aj, denom, H1, out1, E, N);
    elu_kernel        <<<dim3((N * 64 + TB - 1) / TB), b256, 0, stream>>>(out1, N * 64);

    // Layer 2
    gemm2_wmma_kernel<<<dim3((N + 63) / 64), dim3(128), 0, stream>>>(out1, W2, H2, N);
    fill_zero_kernel <<<dim3((N * 40 + TB - 1) / TB), b256, 0, stream>>>(out2, N * 40);
    fill_zero_kernel <<<dim3((N + TB - 1) / TB), b256, 0, stream>>>(deg, N);
    edge2_aggr_kernel<<<dim3((ET + TB - 1) / TB), b256, 0, stream>>>(ei, H2, out2, deg, E, N);
    logsoftmax_kernel<<<dim3((N + TB - 1) / TB), b256, 0, stream>>>(out2, deg, (float*)d_out, N);
}
